// MFN_83081847374141
// MI455X (gfx1250) — matrix operand order repaired and checked by reading, not compiled, not run on hardware
//
#include <hip/hip_runtime.h>
#include <math.h>

// ---------------------------------------------------------------------------
// MFN forward on gfx1250.  GEMMs: V_WMMA_F32_16X16X32_F16, fp16 operands
// (weights/activations pre-converted, K padded to %32==0), fp32 accumulate.
// B tiles are DMA'd into double-buffered LDS with GLOBAL_LOAD_ASYNC_TO_LDS_B128
// (ASYNCcnt), prefetched one chunk ahead; A fragments are two b128 loads.
// ---------------------------------------------------------------------------

typedef __attribute__((ext_vector_type(16))) _Float16 v16h;
typedef __attribute__((ext_vector_type(8)))  _Float16 v8h;
typedef __attribute__((ext_vector_type(8)))  float    v8f;

#define NB 1024   // batch
#define TT 64     // timesteps
#define DX 409    // D_L + D_A + D_V

#define ACT_NONE 0
#define ACT_RELU 1
#define ACT_TANH 2
#define ACT_SIG  3

__device__ __forceinline__ float sigf(float v) { return 1.f / (1.f + __expf(-v)); }

__device__ __forceinline__ float actf(float v, int act) {
  switch (act) {
    case ACT_RELU: return v > 0.f ? v : 0.f;
    case ACT_TANH: return tanhf(v);
    case ACT_SIG:  return sigf(v);
    default:       return v;
  }
}

// async 16-byte global -> LDS copy (per-lane), tracked by ASYNCcnt
__device__ __forceinline__ void async_b128(unsigned lds_off, const _Float16* gsrc) {
  asm volatile("global_load_async_to_lds_b128 %0, %1, off"
               :: "v"(lds_off), "v"(gsrc) : "memory");
}
__device__ __forceinline__ void wait_async0() {
  asm volatile("s_wait_asynccnt 0x0" ::: "memory");
}

// ---------------------------------------------------------------------------
// C/Ch[nrows x mcols] = act( beta*C + A[nrows x K] @ W[mcols x K]^T + bias )
// A fp16 (row stride lda halves, 16B-aligned rows), W fp16 dense (ldw == Kp),
// C fp32 out (nullable), Ch fp16 mirror out (nullable). K % 32 == 0.
// Block 256 thr (8 waves) -> 128x64 tile; 4 WMMA per wave per 32-K chunk.
// ---------------------------------------------------------------------------
__global__ __launch_bounds__(256)
void gemm_wmma_f16(const _Float16* __restrict__ A, int lda,
                   const _Float16* __restrict__ W, int ldw,
                   const float* __restrict__ bias,
                   float* __restrict__ C, _Float16* __restrict__ Ch, int ldc,
                   int mcols, int K, int beta, int act)
{
  __shared__ _Float16 ldsB[2][64 * 32];   // [buf][col*32 + k] fp16, 8 KB

  const int tid  = threadIdx.x;
  const int lane = tid & 31;
  const int wave = tid >> 5;          // 16-row strip within 128-row tile
  const int half = lane >> 4;
  const int l15  = lane & 15;

  const int bm = blockIdx.x;          // 64-col tile
  const int bn = blockIdx.y;          // 128-row tile

  const int arow = bn * 128 + wave * 16 + l15;
  const _Float16* __restrict__ Arow = A + (size_t)arow * lda;

  // cooperative async B: thread -> (col 0..63, 16B k-segment 0..3)
  const int lcol = tid >> 2;          // 0..63
  const int kseg = (tid & 3) << 3;    // halves: 0,8,16,24
  const int gcol = bm * 64 + lcol;
  const int gcolC = (gcol < mcols) ? gcol : (mcols - 1);   // clamp (dup rows ok)
  const _Float16* __restrict__ Wseg = W + (size_t)gcolC * ldw + kseg;

  const unsigned ldsOff0 = (unsigned)(uintptr_t)(&ldsB[0][lcol * 32 + kseg]);
  const unsigned ldsOff1 = (unsigned)(uintptr_t)(&ldsB[1][lcol * 32 + kseg]);

  v8f acc[4] = {v8f{}, v8f{}, v8f{}, v8f{}};

  const int nchunks = K >> 5;
  async_b128(ldsOff0, Wseg);                       // prefetch chunk 0

  for (int i = 0; i < nchunks; ++i) {
    wait_async0();                                  // my chunk-i copy done
    __syncthreads();                                // everyone's copy done
    if (i + 1 < nchunks)                            // prefetch next chunk
      async_b128((i & 1) ? ldsOff0 : ldsOff1, Wseg + (size_t)(i + 1) * 32);

    const int k0 = i << 5;
    // A fragment: two aligned 16B loads, combined per ISA 16-bit A layout
    v8h a0 = *(const v8h*)(Arow + k0 + (half << 3));
    v8h a1 = *(const v8h*)(Arow + k0 + (half << 3) + 16);
    v16h af = __builtin_shufflevector(a0, a1, 0, 1, 2, 3, 4, 5, 6, 7,
                                              8, 9, 10, 11, 12, 13, 14, 15);

    const _Float16* lb = ldsB[i & 1];
    #pragma unroll
    for (int ct = 0; ct < 4; ++ct) {
      v8h b0 = *(const v8h*)&lb[(ct * 16 + l15) * 32 + (half << 3)];
      v8h b1 = *(const v8h*)&lb[(ct * 16 + l15) * 32 + (half << 3) + 16];
      v16h bf = __builtin_shufflevector(b0, b1, 0, 1, 2, 3, 4, 5, 6, 7,
                                                8, 9, 10, 11, 12, 13, 14, 15);
      acc[ct] = __builtin_amdgcn_wmma_f32_16x16x32_f16(
          false, af, false, bf, (short)0, acc[ct], false, false);
    }
  }

  // epilogue
  #pragma unroll
  for (int ct = 0; ct < 4; ++ct) {
    const int col = bm * 64 + ct * 16 + l15;
    if (col < mcols) {
      const float b = bias ? bias[col] : 0.f;
      #pragma unroll
      for (int r = 0; r < 8; ++r) {
        const int row = bn * 128 + wave * 16 + half * 8 + r;
        const size_t idx = (size_t)row * ldc + col;
        float v = acc[ct][r] + b;
        if (beta) v += C[idx];
        v = actf(v, act);
        if (C)  C[idx]  = v;
        if (Ch) Ch[idx] = (_Float16)v;
      }
    }
  }
}

// ---------------------------------------------------------------------------
// Conversion / elementwise kernels
// ---------------------------------------------------------------------------
__global__ void zero_f32(float* __restrict__ p, int n)
{
  int i = blockIdx.x * blockDim.x + threadIdx.x;
  if (i < n) p[i] = 0.f;
}
__global__ void zero_f16(_Float16* __restrict__ p, int n)
{
  int i = blockIdx.x * blockDim.x + threadIdx.x;
  if (i < n) p[i] = (_Float16)0.f;
}

// fp32 [M x K] -> fp16 [M x Kp], zero-padding K..Kp
__global__ void cvt_w(const float* __restrict__ src, _Float16* __restrict__ dst,
                      int M, int K, int Kp)
{
  int i = blockIdx.x * blockDim.x + threadIdx.x;
  if (i >= M * Kp) return;
  int m = i / Kp, k = i - m * Kp;
  dst[i] = (_Float16)((k < K) ? src[(size_t)m * K + k] : 0.f);
}

// split one timestep of x into padded fp16 modality slices
__global__ void cvt_x(const float* __restrict__ xt,
                      _Float16* __restrict__ xl,   // [NB x 320], K=300
                      _Float16* __restrict__ xa,   // [NB x  96], K= 74
                      _Float16* __restrict__ xv)   // [NB x  64], K= 35
{
  int i = blockIdx.x * blockDim.x + threadIdx.x;
  if (i >= NB * 480) return;
  int n = i / 480, j = i - n * 480;
  const float* row = xt + (size_t)n * DX;
  if (j < 320)      xl[(size_t)n * 320 + j] = (_Float16)((j < 300) ? row[j] : 0.f);
  else if (j < 416) { int k = j - 320; xa[(size_t)n * 96 + k] = (_Float16)((k < 74) ? row[300 + k] : 0.f); }
  else              { int k = j - 416; xv[(size_t)n * 64 + k] = (_Float16)((k < 35) ? row[374 + k] : 0.f); }
}

// LSTM gate nonlinearity; writes c (fp32), h (fp16), cStar (fp32 + fp16).
// Hidden layout: [0,512)=L, [512,768)=A, [768,1024)=V.
__global__ void lstm_pointwise(const float* __restrict__ gl,
                               const float* __restrict__ ga,
                               const float* __restrict__ gv,
                               float* __restrict__ c,
                               _Float16* __restrict__ hh,
                               float* __restrict__ cstar,
                               _Float16* __restrict__ cstarh)
{
  int i = blockIdx.x * blockDim.x + threadIdx.x;
  if (i >= NB * 1024) return;
  int n = i >> 10;
  int j = i & 1023;

  const float* G; int H, jl, ld;
  if (j < 512)      { G = gl; H = 512; jl = j;       ld = 2048; }
  else if (j < 768) { G = ga; H = 256; jl = j - 512; ld = 1024; }
  else              { G = gv; H = 256; jl = j - 768; ld = 1024; }

  const float* g = G + (size_t)n * ld;
  float gi = g[jl], gf = g[jl + H], gg = g[jl + 2 * H], go = g[jl + 3 * H];

  float cold = c[i];
  float cnew = sigf(gf) * cold + sigf(gi) * tanhf(gg);
  float hnew = sigf(go) * tanhf(cnew);
  c[i]  = cnew;
  hh[i] = (_Float16)hnew;
  size_t b = (size_t)n * 2048;
  cstar[b + j]         = cold;
  cstar[b + 1024 + j]  = cnew;
  cstarh[b + j]        = (_Float16)cold;
  cstarh[b + 1024 + j] = (_Float16)cnew;
}

// Row softmax over att_o[N x 2048]; both_h = [softmax*cStar (2048) | mem (512)] fp16
__global__ void softmax_attend(const float* __restrict__ att_o,
                               const float* __restrict__ cstar,
                               const float* __restrict__ mem,
                               _Float16* __restrict__ bothh)
{
  const int n = blockIdx.x;
  const int t = threadIdx.x;
  const float* a  = att_o + (size_t)n * 2048;
  const float* cs = cstar + (size_t)n * 2048;
  _Float16* bo = bothh + (size_t)n * 2560;
  __shared__ float red[256];

  float m = -INFINITY;
  for (int j = t; j < 2048; j += 256) m = fmaxf(m, a[j]);
  red[t] = m; __syncthreads();
  for (int s = 128; s > 0; s >>= 1) {
    if (t < s) red[t] = fmaxf(red[t], red[t + s]);
    __syncthreads();
  }
  m = red[0]; __syncthreads();

  float sum = 0.f;
  for (int j = t; j < 2048; j += 256) sum += __expf(a[j] - m);
  red[t] = sum; __syncthreads();
  for (int s = 128; s > 0; s >>= 1) {
    if (t < s) red[t] += red[t + s];
    __syncthreads();
  }
  const float inv = 1.f / red[0];

  for (int j = t; j < 2048; j += 256)
    bo[j] = (_Float16)((__expf(a[j] - m) * inv) * cs[j]);
  for (int j = t; j < 512; j += 256)
    bo[2048 + j] = (_Float16)mem[(size_t)n * 512 + j];
}

__global__ void mem_update(float* __restrict__ mem,
                           const float* __restrict__ g1,
                           const float* __restrict__ g2,
                           const float* __restrict__ chat)
{
  int i = blockIdx.x * blockDim.x + threadIdx.x;
  if (i >= NB * 512) return;
  mem[i] = g1[i] * mem[i] + g2[i] * chat[i];
}

// last_h = [h(1024) | mem(512)] fp16
__global__ void concat_final(const _Float16* __restrict__ hh,
                             const float* __restrict__ mem,
                             _Float16* __restrict__ last)
{
  int i = blockIdx.x * blockDim.x + threadIdx.x;
  if (i >= NB * 1536) return;
  int n = i / 1536;
  int j = i - n * 1536;
  last[i] = (j < 1024) ? hh[(size_t)n * 1024 + j]
                       : (_Float16)mem[(size_t)n * 512 + (j - 1024)];
}

// ---------------------------------------------------------------------------
// Host side
// ---------------------------------------------------------------------------
static inline void gemm(hipStream_t s, const _Float16* A, int lda,
                        const _Float16* W, int ldw, const float* bias,
                        float* C, _Float16* Ch, int ldc,
                        int mcols, int K, int beta, int act)
{
  dim3 grid((mcols + 63) / 64, NB / 128);
  gemm_wmma_f16<<<grid, 256, 0, s>>>(A, lda, W, ldw, bias, C, Ch, ldc,
                                     mcols, K, beta, act);
}

// fp32 workspace offsets (floats)
#define F_C      ((size_t)0)
#define F_MEM    (F_C     + (size_t)NB * 1024)
#define F_GL     (F_MEM   + (size_t)NB * 512)
#define F_GA     (F_GL    + (size_t)NB * 2048)
#define F_GV     (F_GA    + (size_t)NB * 1024)
#define F_CSTAR  (F_GV    + (size_t)NB * 1024)
#define F_ATTO   (F_CSTAR + (size_t)NB * 2048)
#define F_CHAT   (F_ATTO  + (size_t)NB * 2048)
#define F_G1O    (F_CHAT  + (size_t)NB * 512)
#define F_G2O    (F_G1O   + (size_t)NB * 512)
#define F_END    (F_G2O   + (size_t)NB * 512)

// fp16 offsets (halves), relative to fp16 base after fp32 region
#define H_H      ((size_t)0)
#define H_CSTAR  (H_H     + (size_t)NB * 1024)
#define H_ATTH   (H_CSTAR + (size_t)NB * 2048)
#define H_BOTH   (H_ATTH  + (size_t)NB * 1024)
#define H_GH     (H_BOTH  + (size_t)NB * 2560)
#define H_LAST   (H_GH    + (size_t)NB * 1024)
#define H_OUTH   (H_LAST  + (size_t)NB * 1536)
#define H_XL     (H_OUTH  + (size_t)NB * 1024)
#define H_XA     (H_XL    + (size_t)NB * 320)
#define H_XV     (H_XA    + (size_t)NB * 96)
#define H_WIHL   (H_XV    + (size_t)NB * 64)
#define H_WHHL   (H_WIHL  + (size_t)2048 * 320)
#define H_WIHA   (H_WHHL  + (size_t)2048 * 512)
#define H_WHHA   (H_WIHA  + (size_t)1024 * 96)
#define H_WIHV   (H_WHHA  + (size_t)1024 * 256)
#define H_WHHV   (H_WIHV  + (size_t)1024 * 64)
#define H_A1W1   (H_WHHV  + (size_t)1024 * 256)
#define H_A1W2   (H_A1W1  + (size_t)1024 * 2048)
#define H_A2W1   (H_A1W2  + (size_t)2048 * 1024)
#define H_A2W2   (H_A2W1  + (size_t)1024 * 2048)
#define H_G1W1   (H_A2W2  + (size_t)512 * 1024)
#define H_G1W2   (H_G1W1  + (size_t)1024 * 2560)
#define H_G2W1   (H_G1W2  + (size_t)512 * 1024)
#define H_G2W2   (H_G2W1  + (size_t)1024 * 2560)
#define H_OW1    (H_G2W2  + (size_t)512 * 1024)
#define H_OW2    (H_OW1   + (size_t)1024 * 1536)

extern "C" void kernel_launch(void* const* d_in, const int* in_sizes, int n_in,
                              void* d_out, int out_size, void* d_ws, size_t ws_size,
                              hipStream_t stream) {
  const float* x       = (const float*)d_in[0];
  // d_in[1..4] priors unused: reference starts its scan carry from zeros.
  const float* W_ih_l  = (const float*)d_in[5];
  const float* W_hh_l  = (const float*)d_in[6];
  const float* b_ih_l  = (const float*)d_in[7];
  const float* b_hh_l  = (const float*)d_in[8];
  const float* W_ih_a  = (const float*)d_in[9];
  const float* W_hh_a  = (const float*)d_in[10];
  const float* b_ih_a  = (const float*)d_in[11];
  const float* b_hh_a  = (const float*)d_in[12];
  const float* W_ih_v  = (const float*)d_in[13];
  const float* W_hh_v  = (const float*)d_in[14];
  const float* b_ih_v  = (const float*)d_in[15];
  const float* b_hh_v  = (const float*)d_in[16];
  const float* att1_w1 = (const float*)d_in[17];
  const float* att1_b1 = (const float*)d_in[18];
  const float* att1_w2 = (const float*)d_in[19];
  const float* att1_b2 = (const float*)d_in[20];
  const float* att2_w1 = (const float*)d_in[21];
  const float* att2_b1 = (const float*)d_in[22];
  const float* att2_w2 = (const float*)d_in[23];
  const float* att2_b2 = (const float*)d_in[24];
  const float* g1_w1   = (const float*)d_in[25];
  const float* g1_b1   = (const float*)d_in[26];
  const float* g1_w2   = (const float*)d_in[27];
  const float* g1_b2   = (const float*)d_in[28];
  const float* g2_w1   = (const float*)d_in[29];
  const float* g2_b1   = (const float*)d_in[30];
  const float* g2_w2   = (const float*)d_in[31];
  const float* g2_b2   = (const float*)d_in[32];
  const float* out_w1  = (const float*)d_in[33];
  const float* out_b1  = (const float*)d_in[34];
  const float* out_w2  = (const float*)d_in[35];
  const float* out_b2  = (const float*)d_in[36];

  float* ws = (float*)d_ws;
  float* c     = ws + F_C;
  float* mem   = ws + F_MEM;
  float* gl    = ws + F_GL;
  float* ga    = ws + F_GA;
  float* gv    = ws + F_GV;
  float* cstar = ws + F_CSTAR;
  float* atto  = ws + F_ATTO;
  float* chat  = ws + F_CHAT;
  float* g1o   = ws + F_G1O;
  float* g2o   = ws + F_G2O;

  _Float16* hb = (_Float16*)(ws + F_END);
  _Float16* hh      = hb + H_H;
  _Float16* cstarh  = hb + H_CSTAR;
  _Float16* atth    = hb + H_ATTH;
  _Float16* bothh   = hb + H_BOTH;
  _Float16* ghh     = hb + H_GH;
  _Float16* lasth   = hb + H_LAST;
  _Float16* outhh   = hb + H_OUTH;
  _Float16* xlh     = hb + H_XL;
  _Float16* xah     = hb + H_XA;
  _Float16* xvh     = hb + H_XV;
  float* out = (float*)d_out;

  // ---- one-time (per call) weight conversion to padded fp16 ----
  auto cvtw = [&](const float* src, size_t off, int M, int K, int Kp) {
    int n = M * Kp;
    cvt_w<<<(n + 255) / 256, 256, 0, stream>>>(src, hb + off, M, K, Kp);
  };
  cvtw(W_ih_l, H_WIHL, 2048,  300,  320);
  cvtw(W_hh_l, H_WHHL, 2048,  512,  512);
  cvtw(W_ih_a, H_WIHA, 1024,   74,   96);
  cvtw(W_hh_a, H_WHHA, 1024,  256,  256);
  cvtw(W_ih_v, H_WIHV, 1024,   35,   64);
  cvtw(W_hh_v, H_WHHV, 1024,  256,  256);
  cvtw(att1_w1, H_A1W1, 1024, 2048, 2048);
  cvtw(att1_w2, H_A1W2, 2048, 1024, 1024);
  cvtw(att2_w1, H_A2W1, 1024, 2048, 2048);
  cvtw(att2_w2, H_A2W2,  512, 1024, 1024);
  cvtw(g1_w1,   H_G1W1, 1024, 2560, 2560);
  cvtw(g1_w2,   H_G1W2,  512, 1024, 1024);
  cvtw(g2_w1,   H_G2W1, 1024, 2560, 2560);
  cvtw(g2_w2,   H_G2W2,  512, 1024, 1024);
  cvtw(out_w1,  H_OW1,  1024, 1536, 1536);
  cvtw(out_w2,  H_OW2,     1, 1024, 1024);

  // ---- zero initial state: c|mem (fp32, contiguous) and h (fp16) ----
  {
    int n0 = (int)(F_GL - F_C);
    zero_f32<<<(n0 + 255) / 256, 256, 0, stream>>>(ws, n0);
    int n1 = NB * 1024;
    zero_f16<<<(n1 + 255) / 256, 256, 0, stream>>>(hh, n1);
  }

  for (int t = 0; t < TT; ++t) {
    const float* xt = x + (size_t)t * NB * DX;
    cvt_x<<<(NB * 480 + 255) / 256, 256, 0, stream>>>(xt, xlh, xah, xvh);

    // --- LSTM gate GEMMs (x-part, then h-part accumulated) ---
    gemm(stream, xlh,      320,  hb + H_WIHL, 320, b_ih_l, gl, 0, 2048, 2048, 320, 0, ACT_NONE);
    gemm(stream, hh,       1024, hb + H_WHHL, 512, b_hh_l, gl, 0, 2048, 2048, 512, 1, ACT_NONE);
    gemm(stream, xah,      96,   hb + H_WIHA,  96, b_ih_a, ga, 0, 1024, 1024,  96, 0, ACT_NONE);
    gemm(stream, hh + 512, 1024, hb + H_WHHA, 256, b_hh_a, ga, 0, 1024, 1024, 256, 1, ACT_NONE);
    gemm(stream, xvh,      64,   hb + H_WIHV,  64, b_ih_v, gv, 0, 1024, 1024,  64, 0, ACT_NONE);
    gemm(stream, hh + 768, 1024, hb + H_WHHV, 256, b_hh_v, gv, 0, 1024, 1024, 256, 1, ACT_NONE);

    lstm_pointwise<<<(NB * 1024 + 255) / 256, 256, 0, stream>>>(gl, ga, gv, c, hh, cstar, cstarh);

    // --- attention MLP 1: 2048 -> 1024 (relu) -> 2048 ---
    gemm(stream, cstarh, 2048, hb + H_A1W1, 2048, att1_b1, 0, atth, 1024, 1024, 2048, 0, ACT_RELU);
    gemm(stream, atth,   1024, hb + H_A1W2, 1024, att1_b2, atto, 0, 2048, 2048, 1024, 0, ACT_NONE);

    softmax_attend<<<NB, 256, 0, stream>>>(atto, cstar, mem, bothh);

    // --- attention MLP 2: 2048 -> 1024 (relu) -> 512 (tanh) ---
    gemm(stream, bothh, 2560, hb + H_A2W1, 2048, att2_b1, 0, atth, 1024, 1024, 2048, 0, ACT_RELU);
    gemm(stream, atth,  1024, hb + H_A2W2, 1024, att2_b2, chat, 0, 512, 512, 1024, 0, ACT_TANH);

    // --- gamma1 / gamma2: 2560 -> 1024 (relu) -> 512 (sigmoid) ---
    gemm(stream, bothh, 2560, hb + H_G1W1, 2560, g1_b1, 0, ghh, 1024, 1024, 2560, 0, ACT_RELU);
    gemm(stream, ghh,   1024, hb + H_G1W2, 1024, g1_b2, g1o, 0, 512, 512, 1024, 0, ACT_SIG);
    gemm(stream, bothh, 2560, hb + H_G2W1, 2560, g2_b1, 0, ghh, 1024, 1024, 2560, 0, ACT_RELU);
    gemm(stream, ghh,   1024, hb + H_G2W2, 1024, g2_b2, g2o, 0, 512, 512, 1024, 0, ACT_SIG);

    mem_update<<<(NB * 512 + 255) / 256, 256, 0, stream>>>(mem, g1o, g2o, chat);
  }

  // --- readout: [h | mem] -> 1024 (relu) -> 1 ---
  concat_final<<<(NB * 1536 + 255) / 256, 256, 0, stream>>>(hh, mem, lasth);
  gemm(stream, lasth, 1536, hb + H_OW1, 1536, out_b1, 0, outhh, 1024, 1024, 1536, 0, ACT_RELU);
  gemm(stream, outhh, 1024, hb + H_OW2, 1024, out_b2, out, 0, 1, 1, 1024, 0, ACT_NONE);
}
